// GraphSAGE_17824114278988
// MI455X (gfx1250) — matrix operand order repaired and checked by reading, not compiled, not run on hardware
//
#include <hip/hip_runtime.h>
#include <hip/hip_bf16.h>

// ---------------------------------------------------------------------------
// GraphSAGE forward for MI455X (gfx1250, wave32).
//  - dense GEMMs on v_wmma_f32_16x16x32_bf16 (bf16 operands, f32 accumulate)
//  - A/B operands pre-swizzled into WMMA fragment layout -> contiguous,
//    coalesced b128 loads; 2 M-tiles per wave so each B fragment feeds 2 WMMAs
//  - edge aggregation via global_atomic_add_f32 (h/agg fit in 192MB L2)
// ---------------------------------------------------------------------------

typedef __attribute__((ext_vector_type(16))) __bf16 v16bf;
typedef __attribute__((ext_vector_type(8)))  __bf16 v8bf;
typedef __attribute__((ext_vector_type(8)))  float  v8f;

#define F_IN 100
#define KPAD 128   // F_IN padded to K multiple of 32
#define H    64
#define CNUM 18
#define C_PAD 32   // classifier N padded to tile multiple

static __device__ __forceinline__ __bf16 f2bf(float f) {
  // round-to-nearest-even fp32 -> bf16
  unsigned u = __builtin_bit_cast(unsigned, f);
  u += 0x7FFFu + ((u >> 16) & 1u);
  unsigned short s = (unsigned short)(u >> 16);
  return __builtin_bit_cast(__bf16, s);
}

static __device__ __forceinline__ unsigned pack2bf(float a, float b) {
  unsigned lo = __builtin_bit_cast(unsigned short, f2bf(a));
  unsigned hi = __builtin_bit_cast(unsigned short, f2bf(b));
  return lo | (hi << 16);
}

// --------------------------- utility kernels -------------------------------

__global__ void zero_f32(float* __restrict__ p, long n) {
  long i = (long)blockIdx.x * blockDim.x + threadIdx.x;
  if (i < n) p[i] = 0.0f;
}

// Weight fp32 [Ksrc x Nsrc] -> bf16 in B-fragment order:
//   dst[((kc*NTILES+nt)*32 + lane)*16 + i] = W[kc*32 + (i&8)*2 + (lane>>4)*8 + (i&7)][nt*16 + (lane&15)]
// (zero-padded outside Ksrc/Nsrc). Each fragment = contiguous 32B per lane.
__global__ void swizzle_weight_bf16(const float* __restrict__ src, __bf16* __restrict__ dst,
                                    int Ksrc, int Nsrc, int ntiles, long n) {
  long idx = (long)blockIdx.x * blockDim.x + threadIdx.x;
  if (idx >= n) return;
  int i    = (int)(idx & 15);
  int lane = (int)((idx >> 4) & 31);
  int frag = (int)(idx >> 9);
  int kc = frag / ntiles, nt = frag % ntiles;
  int K = kc * 32 + (i & 8) * 2 + (lane >> 4) * 8 + (i & 7);
  int N = nt * 16 + (lane & 15);
  float v = (K < Ksrc && N < Nsrc) ? src[(long)K * Nsrc + N] : 0.0f;
  dst[idx] = f2bf(v);
}

// x fp32 [Nnodes x F_IN] -> bf16 in A-fragment order (K padded to KPAD):
//   dst[((tile*KC+kc)*32 + lane)*16 + i] = x[tile*16 + (lane&15)]
//                                           [kc*32 + (i&8)*2 + (lane>>4)*8 + (i&7)]
__global__ void swizzle_x_bf16(const float* __restrict__ src, __bf16* __restrict__ dst,
                               int fin, long n) {
  long idx = (long)blockIdx.x * blockDim.x + threadIdx.x;
  if (idx >= n) return;
  int i    = (int)(idx & 15);
  int lane = (int)((idx >> 4) & 31);
  long frag = idx >> 9;
  int kc   = (int)(frag & (KPAD / 32 - 1));   // KC = 4 chunks
  long tile = frag >> 2;
  int row = (int)(tile * 16 + (lane & 15));
  int col = kc * 32 + (i & 8) * 2 + (lane >> 4) * 8 + (i & 7);
  float v = (col < fin) ? src[(long)row * fin + col] : 0.0f;
  dst[idx] = f2bf(v);
}

__global__ void compute_deg(const int* __restrict__ dst, float* __restrict__ deg, int E) {
  int e = blockIdx.x * blockDim.x + threadIdx.x;
  if (e < E) atomicAdd(&deg[dst[e]], 1.0f);
}

// one thread per (edge, 4-feature group): msg = w * h[src]; agg[dst] += msg
__global__ void scatter_edges(const int* __restrict__ src, const int* __restrict__ dst,
                              const float* __restrict__ ew, const float* __restrict__ h,
                              float* __restrict__ agg, long total) {
  long idx = (long)blockIdx.x * blockDim.x + threadIdx.x;
  if (idx >= total) return;
  int e = (int)(idx >> 4);
  int f = (int)(idx & 15) << 2;
  float w = ew ? ew[e] : 1.0f;
  const float4 v = *(const float4*)(h + (size_t)src[e] * H + f);
  float* a = agg + (size_t)dst[e] * H + f;
  atomicAdd(a + 0, w * v.x);
  atomicAdd(a + 1, w * v.y);
  atomicAdd(a + 2, w * v.z);
  atomicAdd(a + 3, w * v.w);
}

// --------------------------- WMMA fragment helpers -------------------------
// A (16x32 bf16) per-lane: lane l -> row m=l&15, half=l>>4;
//   elem i -> K = kc*32 + (i&8)*2 + half*8 + (i&7)  (two contiguous 16B runs)
// B (32x16 bf16): lane l -> col n=l&15; elem i -> K = kc*32 + (i&8)*2 + half*8 + (i&7) (same as A).
// C/D (16x16 f32): VGPR r, lanes 0-15 -> row r, lanes 16-31 -> row r+8; col = lane&15.

static __device__ __forceinline__ v16bf load_frag32B(const __bf16* __restrict__ base,
                                                     int frag, int lane) {
  return *(const v16bf*)(base + ((size_t)frag * 32 + lane) * 16);
}

// Embedding: h = relu(x @ W + b); x pre-swizzled A-format, W pre-swizzled B-format.
// One wave handles 2 node tiles (32 nodes) so each B fragment feeds 2 WMMAs.
__global__ __launch_bounds__(32)
void embed_wmma(const __bf16* __restrict__ xb, const __bf16* __restrict__ Wb,
                const float* __restrict__ bias, float* __restrict__ h_out) {
  const int tp   = blockIdx.x;          // tile pair
  const int lane = threadIdx.x;
  const int m = lane & 15, half = lane >> 4;
  const int KC = KPAD / 32, NT = 4;

  v8f acc[2][NT];
#pragma unroll
  for (int nt = 0; nt < NT; ++nt) {
    float b = bias[nt * 16 + m];
#pragma unroll
    for (int r = 0; r < 8; ++r) { acc[0][nt][r] = b; acc[1][nt][r] = b; }
  }

#pragma unroll
  for (int kc = 0; kc < KC; ++kc) {
    v16bf a0 = load_frag32B(xb, (tp * 2 + 0) * KC + kc, lane);
    v16bf a1 = load_frag32B(xb, (tp * 2 + 1) * KC + kc, lane);
#pragma unroll
    for (int nt = 0; nt < NT; ++nt) {
      v16bf bf = load_frag32B(Wb, kc * NT + nt, lane);
      acc[0][nt] = __builtin_amdgcn_wmma_f32_16x16x32_bf16(false, a0, false, bf,
                                                           (short)0, acc[0][nt], false, false);
      acc[1][nt] = __builtin_amdgcn_wmma_f32_16x16x32_bf16(false, a1, false, bf,
                                                           (short)0, acc[1][nt], false, false);
    }
  }

#pragma unroll
  for (int t = 0; t < 2; ++t)
#pragma unroll
    for (int r = 0; r < 8; ++r) {
      int node = tp * 32 + t * 16 + r + 8 * half;
#pragma unroll
      for (int nt = 0; nt < NT; ++nt)
        h_out[(size_t)node * H + nt * 16 + m] = fmaxf(acc[t][nt][r], 0.0f);
    }
}

// SAGE update: out = L2norm( (agg/deg) @ Wl + bl + h @ Wr ), optional relu.
// Weights pre-swizzled B-format with NTILES tiles; ncols_out = true width (64/18).
// One wave handles 32 nodes (2 M-tiles) sharing each B fragment.
template <int NTILES, bool RELU>
__global__ __launch_bounds__(32)
void sage_update_wmma(const float* __restrict__ agg, const float* __restrict__ deg,
                      const float* __restrict__ h_in,
                      const __bf16* __restrict__ Wl, const __bf16* __restrict__ Wr,
                      const float* __restrict__ bl, float* __restrict__ h_out,
                      int ncols_out) {
  const int tp   = blockIdx.x;          // 32-node tile pair
  const int lane = threadIdx.x;
  const int m = lane & 15, half = lane >> 4;

  __shared__ __bf16 sMean[32][H];
  __shared__ __bf16 sRoot[32][H];

  // vectorized staging: float4 global loads -> 2x packed-bf16 dwords into LDS
  for (int q = lane; q < 32 * H / 4; q += 32) {
    int r = q >> 4;                 // node within the 32-node pair
    int c = (q & 15) << 2;          // feature column, step 4
    int node = tp * 32 + r;
    float invd = 1.0f / fmaxf(deg[node], 1.0f);
    float4 va = *(const float4*)(agg + (size_t)node * H + c);
    float4 vh = *(const float4*)(h_in + (size_t)node * H + c);
    uint2 pm, ph;
    pm.x = pack2bf(va.x * invd, va.y * invd);
    pm.y = pack2bf(va.z * invd, va.w * invd);
    ph.x = pack2bf(vh.x, vh.y);
    ph.y = pack2bf(vh.z, vh.w);
    *(uint2*)&sMean[r][c] = pm;
    *(uint2*)&sRoot[r][c] = ph;
  }
  __syncthreads();

  // A fragments from LDS: two contiguous 16B runs per lane per K-chunk
  v16bf aM[2][2], aR[2][2];         // [tile][kc]
#pragma unroll
  for (int t = 0; t < 2; ++t)
#pragma unroll
    for (int kc = 0; kc < 2; ++kc) {
      int row  = t * 16 + m;
      int base = kc * 32 + half * 8;
      v8bf lo = *(const v8bf*)&sMean[row][base];
      v8bf hi = *(const v8bf*)&sMean[row][base + 16];
#pragma unroll
      for (int i = 0; i < 8; ++i) { aM[t][kc][i] = lo[i]; aM[t][kc][i + 8] = hi[i]; }
      lo = *(const v8bf*)&sRoot[row][base];
      hi = *(const v8bf*)&sRoot[row][base + 16];
#pragma unroll
      for (int i = 0; i < 8; ++i) { aR[t][kc][i] = lo[i]; aR[t][kc][i + 8] = hi[i]; }
    }

  v8f acc[2][NTILES];
#pragma unroll
  for (int nt = 0; nt < NTILES; ++nt) {
    int N = nt * 16 + m;
    float b = (N < ncols_out) ? bl[N] : 0.0f;   // padded cols stay exactly 0
#pragma unroll
    for (int r = 0; r < 8; ++r) { acc[0][nt][r] = b; acc[1][nt][r] = b; }
  }

#pragma unroll
  for (int kc = 0; kc < 2; ++kc) {
#pragma unroll
    for (int nt = 0; nt < NTILES; ++nt) {
      v16bf bL = load_frag32B(Wl, kc * NTILES + nt, lane);
      v16bf bR = load_frag32B(Wr, kc * NTILES + nt, lane);
#pragma unroll
      for (int t = 0; t < 2; ++t) {
        acc[t][nt] = __builtin_amdgcn_wmma_f32_16x16x32_bf16(false, aM[t][kc], false, bL,
                                                             (short)0, acc[t][nt], false, false);
        acc[t][nt] = __builtin_amdgcn_wmma_f32_16x16x32_bf16(false, aR[t][kc], false, bR,
                                                             (short)0, acc[t][nt], false, false);
      }
    }
  }

  // per-row L2 norm: row = r + 8*half lives in VGPR r across the 16 lanes of a half
#pragma unroll
  for (int t = 0; t < 2; ++t) {
    float inv[8];
#pragma unroll
    for (int r = 0; r < 8; ++r) {
      float p = 0.0f;
#pragma unroll
      for (int nt = 0; nt < NTILES; ++nt) p += acc[t][nt][r] * acc[t][nt][r];
      p += __shfl_xor(p, 1, 32);
      p += __shfl_xor(p, 2, 32);
      p += __shfl_xor(p, 4, 32);
      p += __shfl_xor(p, 8, 32);
      inv[r] = 1.0f / fmaxf(sqrtf(p), 1e-12f);
    }
#pragma unroll
    for (int r = 0; r < 8; ++r) {
      int node = tp * 32 + t * 16 + r + 8 * half;
#pragma unroll
      for (int nt = 0; nt < NTILES; ++nt) {
        int N = nt * 16 + m;
        float v = acc[t][nt][r] * inv[r];
        if (RELU) v = fmaxf(v, 0.0f);
        if (N < ncols_out) h_out[(size_t)node * ncols_out + N] = v;
      }
    }
  }
}

// ------------------------------- launcher ----------------------------------

extern "C" void kernel_launch(void* const* d_in, const int* in_sizes, int n_in,
                              void* d_out, int out_size, void* d_ws, size_t ws_size,
                              hipStream_t stream) {
  const float* x    = (const float*)d_in[0];
  const int*   ei   = (const int*)d_in[1];
  const float* ew   = (const float*)d_in[2];
  const float* embW = (const float*)d_in[3];
  const float* embB = (const float*)d_in[4];
  const float* Wlf[4] = {(const float*)d_in[5], (const float*)d_in[8],
                         (const float*)d_in[11], (const float*)d_in[14]};
  const float* blf[4] = {(const float*)d_in[6], (const float*)d_in[9],
                         (const float*)d_in[12], (const float*)d_in[15]};
  const float* Wrf[4] = {(const float*)d_in[7], (const float*)d_in[10],
                         (const float*)d_in[13], (const float*)d_in[16]};

  const int N = in_sizes[0] / F_IN;
  const int E = in_sizes[2];
  const int* src = ei;
  const int* dst = ei + E;

  // workspace carve-up (256B aligned)
  char* ws = (char*)d_ws;
  size_t off = 0;
  auto carve = [&](size_t bytes) -> void* {
    void* p = ws + off;
    off = (off + bytes + 255) & ~(size_t)255;
    return p;
  };
  __bf16* xb    = (__bf16*)carve((size_t)N * KPAD * sizeof(__bf16));     // A-format
  __bf16* embWb = (__bf16*)carve((size_t)KPAD * H * sizeof(__bf16));     // B-format
  __bf16* Wlb[4], *Wrb[4];
  for (int l = 0; l < 3; ++l) {
    Wlb[l] = (__bf16*)carve((size_t)H * H * sizeof(__bf16));
    Wrb[l] = (__bf16*)carve((size_t)H * H * sizeof(__bf16));
  }
  Wlb[3] = (__bf16*)carve((size_t)H * C_PAD * sizeof(__bf16));
  Wrb[3] = (__bf16*)carve((size_t)H * C_PAD * sizeof(__bf16));
  float* h_a = (float*)carve((size_t)N * H * sizeof(float));
  float* h_b = (float*)carve((size_t)N * H * sizeof(float));
  float* agg = (float*)carve((size_t)N * H * sizeof(float));
  float* deg = (float*)carve((size_t)N * sizeof(float));

  const int TPB = 256;
  auto blocks = [](long n, int t) { return (unsigned)((n + t - 1) / t); };

  // 1) bf16 conversion + fragment swizzle (zero padded)
  long nxb = (long)N * KPAD;
  swizzle_x_bf16<<<blocks(nxb, TPB), TPB, 0, stream>>>(x, xb, F_IN, nxb);
  swizzle_weight_bf16<<<blocks((long)KPAD * H, TPB), TPB, 0, stream>>>(
      embW, embWb, F_IN, H, 4, (long)KPAD * H);
  for (int l = 0; l < 3; ++l) {
    swizzle_weight_bf16<<<blocks((long)H * H, TPB), TPB, 0, stream>>>(
        Wlf[l], Wlb[l], H, H, 4, (long)H * H);
    swizzle_weight_bf16<<<blocks((long)H * H, TPB), TPB, 0, stream>>>(
        Wrf[l], Wrb[l], H, H, 4, (long)H * H);
  }
  swizzle_weight_bf16<<<blocks((long)H * C_PAD, TPB), TPB, 0, stream>>>(
      Wlf[3], Wlb[3], H, CNUM, 2, (long)H * C_PAD);
  swizzle_weight_bf16<<<blocks((long)H * C_PAD, TPB), TPB, 0, stream>>>(
      Wrf[3], Wrb[3], H, CNUM, 2, (long)H * C_PAD);

  // 2) degree (same for all layers: unweighted ones over dst)
  zero_f32<<<blocks(N, TPB), TPB, 0, stream>>>(deg, N);
  compute_deg<<<blocks(E, TPB), TPB, 0, stream>>>(dst, deg, E);

  // 3) embedding GEMM + relu  -> h_a   (32 nodes per wave)
  embed_wmma<<<N / 32, 32, 0, stream>>>(xb, embWb, embB, h_a);

  // 4) three hidden SAGE layers (weighted messages, relu)
  const long scat_total = (long)E * 16;
  float* h_cur = h_a;
  float* h_nxt = h_b;
  for (int l = 0; l < 3; ++l) {
    zero_f32<<<blocks((long)N * H, TPB), TPB, 0, stream>>>(agg, (long)N * H);
    scatter_edges<<<blocks(scat_total, TPB), TPB, 0, stream>>>(src, dst, ew, h_cur, agg, scat_total);
    sage_update_wmma<4, true><<<N / 32, 32, 0, stream>>>(agg, deg, h_cur, Wlb[l], Wrb[l],
                                                         blf[l], h_nxt, H);
    float* t = h_cur; h_cur = h_nxt; h_nxt = t;
  }

  // 5) classifier layer (unweighted messages, normalize, no relu) -> d_out
  zero_f32<<<blocks((long)N * H, TPB), TPB, 0, stream>>>(agg, (long)N * H);
  scatter_edges<<<blocks(scat_total, TPB), TPB, 0, stream>>>(src, dst, nullptr, h_cur, agg, scat_total);
  sage_update_wmma<2, false><<<N / 32, 32, 0, stream>>>(agg, deg, h_cur, Wlb[3], Wrb[3],
                                                        blf[3], (float*)d_out, CNUM);

  (void)n_in; (void)out_size; (void)ws_size;
}
